// RBF_Conv2d_1769526526667
// MI455X (gfx1250) — matrix operand order repaired and checked by reading, not compiled, not run on hardware
//
#include <hip/hip_runtime.h>
#include <hip/hip_bf16.h>
#include <stdint.h>

typedef __attribute__((ext_vector_type(16))) __bf16 v16bf;
typedef __attribute__((ext_vector_type(8)))  float  v8f;
typedef __attribute__((ext_vector_type(4)))  int    v4i;
typedef __attribute__((ext_vector_type(8)))  int    v8i;

#define C_IN   64
#define HW     56
#define L_IMG  (HW * HW)          // 3136
#define O_OUT  128
#define KDIM   576
#define M_TILE 64
#define KB_STEPS 18               // 576 / 32
#define TILES_PER_IMG (L_IMG / M_TILE)   // 49 (exact)
#define BPACK_ELEMS (KB_STEPS * 8 * 32 * 16)   // 73728 bf16

__device__ __forceinline__ unsigned short f2bf(float f) {
  unsigned u = __float_as_uint(f);
  u += 0x7FFFu + ((u >> 16) & 1u);       // round-to-nearest-even
  return (unsigned short)(u >> 16);
}

// ---- prep: pack weights as bf16 in WMMA B-fragment order [kb][nt][lane][e] ----
__global__ void pack_weights(const float* __restrict__ w,
                             unsigned short* __restrict__ bh) {
  int idx = blockIdx.x * blockDim.x + threadIdx.x;
  if (idx >= BPACK_ELEMS) return;
  int e    = idx & 15;
  int lane = (idx >> 4) & 31;
  int kbnt = idx >> 9;
  int kb   = kbnt >> 3;
  int nt   = kbnt & 7;
  int k = kb * 32 + (lane >> 4) * 16 + e;
  int n = nt * 16 + (lane & 15);
  bh[idx] = f2bf(w[n * KDIM + k]);
}

__global__ void weight_norms(const float* __restrict__ w, float* __restrict__ w2) {
  int o = threadIdx.x;
  if (o >= O_OUT) return;
  float s = 0.f;
  for (int k = 0; k < KDIM; ++k) { float v = w[o * KDIM + k]; s = fmaf(v, v, s); }
  w2[o] = s;
}

// ---- main: 64(M) x 128(N) tile per workgroup, K=576 via bf16 WMMA ----
__global__ __launch_bounds__(256)
void rbf_conv_wmma(const float* __restrict__ x,
                   const unsigned short* __restrict__ bpack,
                   const float* __restrict__ w2g,
                   float* __restrict__ out) {
  union Sm {
    float xs[C_IN * 4 * HW];             // fp32 slab written by TDM (max 4 rows)
    float res[M_TILE * 129];             // pw results, padded stride
  };
  __shared__ Sm sm;
  __shared__ __attribute__((aligned(16))) unsigned short xpad[C_IN * 4 * 58]; // bf16, zero-padded
  __shared__ __attribute__((aligned(16))) int offtab[KDIM];  // im2col LDS offsets per k
  __shared__ float Spad[4 * 58];         // per-(row,col) channel sum of squares
  __shared__ float p2s[M_TILE];
  __shared__ float w2s[O_OUT];

  const int tid  = threadIdx.x;
  const int b    = blockIdx.x / TILES_PER_IMG;
  const int tile = blockIdx.x % TILES_PER_IMG;
  const int l0   = tile * M_TILE;
  const int h0   = l0 / HW;

  const int ihs = (h0 > 0) ? (h0 - 1) : 0;          // first valid input row
  const int ihe = (h0 + 2 < HW) ? (h0 + 2) : (HW - 1);
  const int nv  = ihe - ihs + 1;                    // 3 or 4 staged rows

  const float* xb = x + (size_t)b * C_IN * L_IMG;

  // ---- TDM: one tensor_load_to_lds moves the whole 56 x nv x 64 fp32 tile ----
  if ((tid >> 5) == 0) {
    unsigned long long ga = (unsigned long long)(uintptr_t)(xb + (size_t)ihs * HW);
    v4i g0 = {0, 0, 0, 0};
    g0[0] = 1;                                        // count=1 (valid user D#)
    g0[1] = (int)(unsigned)(uintptr_t)(&sm.xs[0]);    // lds_addr
    g0[2] = (int)(unsigned)(ga & 0xFFFFFFFFull);      // global_addr[31:0]
    g0[3] = (int)(((ga >> 32) & 0x1FFFFFFull) | 0x80000000u); // addr[56:32] | type=2
    v8i g1 = {0, 0, 0, 0, 0, 0, 0, 0};
    g1[0] = 0x00020000;                 // workgroup_mask=0, data_size=4B
    g1[1] = (HW << 16);                 // tensor_dim0 = 56   (bits 79:48)
    g1[2] = (HW << 16);                 // tensor_dim1 = 56   (bits 111:80)
    g1[3] = (HW << 16);                 // tile_dim0  = 56    (bits 127:112)
    g1[4] = nv | (C_IN << 16);          // tile_dim1 = nv, tile_dim2 = 64
    g1[5] = HW;                         // tensor_dim0_stride = 56
    g1[6] = (L_IMG << 16);              // tensor_dim1_stride = 3136 (bits 223:208)
    g1[7] = 0;
    v4i g2 = {C_IN, 0, 0, 0};           // tensor_dim2 = 64; no dim3
    v4i g3 = {0, 0, 0, 0};
    asm volatile("tensor_load_to_lds %0, %1, %2, %3"
                 :: "s"(g0), "s"(g1), "s"(g2), "s"(g3) : "memory");
  }

  // ---- async global->LDS for the ||w||^2 vector ----
  if (tid < O_OUT) {
    unsigned loff = (unsigned)(uintptr_t)(&w2s[tid]);
    const float* gp = w2g + tid;
    asm volatile("global_load_async_to_lds_b32 %0, %1, off"
                 :: "v"(loff), "v"(gp) : "memory");
  }

  // ---- independent setup while DMA is in flight ----
  for (int k = tid; k < KDIM; k += 256) {            // im2col offset table
    int c  = k / 9;
    int rr = k - c * 9;
    int kh = rr / 3;
    int kw = rr - kh * 3;
    offtab[k] = c * (4 * 58) + kh * 58 + kw;
  }
  if (tid < 4 * 58) Spad[tid] = 0.f;

  __builtin_amdgcn_s_wait_tensorcnt(0);
  asm volatile("s_wait_asynccnt 0" ::: "memory");
  __syncthreads();

  // launder: sm.xs was written behind the compiler's back by the TDM
  const float* xsl = sm.xs;
  asm volatile("" : "+v"(xsl) :: "memory");

  // ---- phase 2a: build zero-padded bf16 tile; thread t <-> (c, hr) ----
  {
    int c  = tid >> 2;
    int hr = tid & 3;
    int ih = h0 - 1 + hr;
    unsigned short* xr = &xpad[tid * 58];
    if ((unsigned)ih < (unsigned)HW) {
      const float* src = xsl + (c * nv + (ih - ihs)) * HW;
      xr[0] = 0; xr[57] = 0;
      for (int iw = 0; iw < HW; ++iw) xr[1 + iw] = f2bf(src[iw]);
    } else {
      for (int i = 0; i < 58; ++i) xr[i] = 0;
    }
  }
  // ---- phase 2b: channel sum-of-squares S[hr][iw]; thread t <-> (hr, iw) ----
  if (tid < 4 * HW) {
    int hr = tid / HW;
    int iw = tid - hr * HW;
    int ih = h0 - 1 + hr;
    if ((unsigned)ih < (unsigned)HW) {
      float s = 0.f;
      int base = (ih - ihs) * HW + iw;
      for (int c = 0; c < C_IN; ++c) {
        float v = xsl[c * (nv * HW) + base];
        s = fmaf(v, v, s);
      }
      Spad[hr * 58 + 1 + iw] = s;
    }
  }
  __syncthreads();

  // ---- p2 per output position: 3x3 window over S ----
  if (tid < M_TILE) {
    int l = l0 + tid;
    int h = l / HW;
    int w = l - h * HW;
    int hr = h - h0;                 // window rows hr..hr+2 in Spad
    float s = 0.f;
#pragma unroll
    for (int dh = 0; dh < 3; ++dh)
#pragma unroll
      for (int dw = 0; dw < 3; ++dw)
        s += Spad[(hr + dh) * 58 + w + dw];
    p2s[tid] = s;
  }

  const int wid  = tid >> 5;
  const int lane = tid & 31;
  const int lg   = lane >> 4;
  const int lr   = lane & 15;
  const int m    = wid & 3;
  const int ng   = wid >> 2;

  const int row = m * 16 + lr;
  const int l   = l0 + row;
  const int h   = l / HW;
  const int w   = l - h * HW;
  const int abase = (h - h0) * 58 + w;   // per-lane im2col base offset

  v8f acc0 = {}, acc1 = {}, acc2 = {}, acc3 = {};
  const v16bf* bp = (const v16bf*)bpack;
  const unsigned short* xp = xpad;

  for (int kb = 0; kb < KB_STEPS; ++kb) {
    const int kbase = kb * 32 + lg * 16;
    int4 ta = *(const int4*)&offtab[kbase];
    int4 tb = *(const int4*)&offtab[kbase + 4];
    int4 tc = *(const int4*)&offtab[kbase + 8];
    int4 td = *(const int4*)&offtab[kbase + 12];
    union { v16bf v; unsigned u[8]; } A;
    A.u[0] = (unsigned)xp[abase + ta.x] | ((unsigned)xp[abase + ta.y] << 16);
    A.u[1] = (unsigned)xp[abase + ta.z] | ((unsigned)xp[abase + ta.w] << 16);
    A.u[2] = (unsigned)xp[abase + tb.x] | ((unsigned)xp[abase + tb.y] << 16);
    A.u[3] = (unsigned)xp[abase + tb.z] | ((unsigned)xp[abase + tb.w] << 16);
    A.u[4] = (unsigned)xp[abase + tc.x] | ((unsigned)xp[abase + tc.y] << 16);
    A.u[5] = (unsigned)xp[abase + tc.z] | ((unsigned)xp[abase + tc.w] << 16);
    A.u[6] = (unsigned)xp[abase + td.x] | ((unsigned)xp[abase + td.y] << 16);
    A.u[7] = (unsigned)xp[abase + td.z] | ((unsigned)xp[abase + td.w] << 16);

    int base = (kb * 8 + ng * 4) * 32 + lane;
    if (kb + 1 < KB_STEPS)
      __builtin_prefetch((const void*)&bp[base + 8 * 32], 0, 1);
    v16bf b0 = bp[base + 0 * 32];
    v16bf b1 = bp[base + 1 * 32];
    v16bf b2 = bp[base + 2 * 32];
    v16bf b3 = bp[base + 3 * 32];
    acc0 = __builtin_amdgcn_wmma_f32_16x16x32_bf16(false, A.v, false, b0, (short)0, acc0, false, false);
    acc1 = __builtin_amdgcn_wmma_f32_16x16x32_bf16(false, A.v, false, b1, (short)0, acc1, false, false);
    acc2 = __builtin_amdgcn_wmma_f32_16x16x32_bf16(false, A.v, false, b2, (short)0, acc2, false, false);
    acc3 = __builtin_amdgcn_wmma_f32_16x16x32_bf16(false, A.v, false, b3, (short)0, acc3, false, false);
  }

  __syncthreads();                       // done with sm.xs; reuse as sm.res

  // C layout: VGPR i -> M = i + 8*lg, N = lr (within 16x16 subtile)
#pragma unroll
  for (int i = 0; i < 8; ++i) {
    int rrow = m * 16 + i + 8 * lg;
    int ob   = (ng * 4) * 16 + lr;
    sm.res[rrow * 129 + ob +  0] = acc0[i];
    sm.res[rrow * 129 + ob + 16] = acc1[i];
    sm.res[rrow * 129 + ob + 32] = acc2[i];
    sm.res[rrow * 129 + ob + 48] = acc3[i];
  }
  __syncthreads();

  // coalesced epilogue: out[b][o][l0+li] = sqrt(max(p2 + w2 - 2*pw, 0))
  float* outb = out + (size_t)b * O_OUT * L_IMG + l0;
  for (int idx = tid; idx < M_TILE * O_OUT; idx += 256) {
    int o  = idx >> 6;
    int li = idx & 63;
    float pw = sm.res[li * 129 + o];
    float d2 = p2s[li] + w2s[o] - 2.0f * pw;
    outb[(size_t)o * L_IMG + li] = sqrtf(fmaxf(d2, 0.f));
  }
}

extern "C" void kernel_launch(void* const* d_in, const int* in_sizes, int n_in,
                              void* d_out, int out_size, void* d_ws, size_t ws_size,
                              hipStream_t stream) {
  (void)in_sizes; (void)n_in; (void)out_size; (void)ws_size;
  const float* x  = (const float*)d_in[0];
  const float* wt = (const float*)d_in[1];
  float* out = (float*)d_out;

  unsigned short* bpack = (unsigned short*)d_ws;
  float* w2 = (float*)((char*)d_ws + (size_t)BPACK_ELEMS * sizeof(unsigned short));

  pack_weights<<<(BPACK_ELEMS + 255) / 256, 256, 0, stream>>>(wt, bpack);
  weight_norms<<<1, 128, 0, stream>>>(wt, w2);
  rbf_conv_wmma<<<16 * TILES_PER_IMG, 256, 0, stream>>>(x, bpack, w2, out);
}
